// FlashSelfAttention_1279900254926
// MI455X (gfx1250) — matrix operand order repaired and checked by reading, not compiled, not run on hardware
//
#include <hip/hip_runtime.h>
#include <hip/hip_bf16.h>

typedef unsigned short u16;
typedef __attribute__((ext_vector_type(16))) __bf16 v16bf;
typedef __attribute__((ext_vector_type(8)))  float  v8f;

union Frag16 { v16bf v; uint4 q[2]; };
union U4E8   { uint4 u; u16 e[8]; };

__device__ __forceinline__ u16 f2bf(float f) {
  unsigned u = __float_as_uint(f);
  u += 0x7FFFu + ((u >> 16) & 1u);            // round-to-nearest-even
  return (u16)(u >> 16);
}
__device__ __forceinline__ float bf2f(u16 h) {
  return __uint_as_float(((unsigned)h) << 16);
}

// Raw LDS byte offset of a __shared__ object (generic -> addrspace(3) -> int)
__device__ __forceinline__ unsigned lds_off_u(const void* p) {
  return (unsigned)(unsigned long long)(const __attribute__((address_space(3))) char*)p;
}

// ---------------------------------------------------------------------------
// f32 -> bf16 conversion, 8 elements per thread
// ---------------------------------------------------------------------------
__global__ void cvt_f32_bf16(const float* __restrict__ src, u16* __restrict__ dst, long n8) {
  long i = (long)blockIdx.x * blockDim.x + threadIdx.x;
  if (i >= n8) return;
  const float4* s = ((const float4*)src) + i * 2;
  float4 a = s[0], b = s[1];
  uint4 o;
  o.x = (unsigned)f2bf(a.x) | ((unsigned)f2bf(a.y) << 16);
  o.y = (unsigned)f2bf(a.z) | ((unsigned)f2bf(a.w) << 16);
  o.z = (unsigned)f2bf(b.x) | ((unsigned)f2bf(b.y) << 16);
  o.w = (unsigned)f2bf(b.z) | ((unsigned)f2bf(b.w) << 16);
  ((uint4*)dst)[i] = o;
}

// ---------------------------------------------------------------------------
// bf16 GEMM: C[M,N] = A[M,K] * B[K,N]. Block tile 128x128x32, 256 threads =
// 8 waves, 4x2 WMMA accumulators per wave. Register double-buffer hides HBM
// latency behind the WMMA chain.
// ---------------------------------------------------------------------------
template <bool OUT_F32>
__global__ __launch_bounds__(256) void gemm_bf16(const u16* __restrict__ A,
                                                 const u16* __restrict__ B,
                                                 void* __restrict__ Cout,
                                                 int M, int N, int K) {
  __shared__ __align__(16) u16 As[128][56];   // 112B rows: 16B-aligned, bank-rotated
  __shared__ __align__(16) u16 Bt[128][56];   // B transposed: Bt[n][k]

  const int tid  = threadIdx.x;
  const int lane = tid & 31;
  const int wid  = tid >> 5;
  const int hl   = lane >> 4;
  const int ln   = lane & 15;
  const int wr   = wid >> 2;        // 0..1 : 64-row strip
  const int wc   = wid & 3;         // 0..3 : 32-col strip
  const int m0   = blockIdx.y * 128;
  const int n0   = blockIdx.x * 128;

  v8f acc[4][2];
#pragma unroll
  for (int i = 0; i < 4; i++)
#pragma unroll
    for (int j = 0; j < 2; j++)
      acc[i][j] = (v8f){0.f, 0.f, 0.f, 0.f, 0.f, 0.f, 0.f, 0.f};

  int aRow[2], aC8[2], bRow[2], bN8[2];
#pragma unroll
  for (int i = 0; i < 2; i++) {
    int idx = tid * 2 + i;          // 0..511 uint4 slots
    aRow[i] = idx >> 2;  aC8[i] = (idx & 3) * 8;
    bRow[i] = idx >> 4;  bN8[i] = (idx & 15) * 8;
  }

  // Prologue: fetch K-tile 0 into registers
  uint4 aReg[2], bReg[2];
#pragma unroll
  for (int i = 0; i < 2; i++) {
    aReg[i] = *(const uint4*)(A + (long)(m0 + aRow[i]) * K + aC8[i]);
    bReg[i] = *(const uint4*)(B + (long)bRow[i] * N + n0 + bN8[i]);
  }

  for (int kk = 0; kk < K; kk += 32) {
    // Commit current tile registers -> LDS
#pragma unroll
    for (int i = 0; i < 2; i++) {
      *(uint4*)(&As[aRow[i]][aC8[i]]) = aReg[i];
      U4E8 t; t.u = bReg[i];
#pragma unroll
      for (int j = 0; j < 8; j++) Bt[bN8[i] + j][bRow[i]] = t.e[j];
    }
    __syncthreads();

    // Prefetch next K-tile into registers (overlaps with WMMA below)
    int kn = kk + 32;
    if (kn < K) {
#pragma unroll
      for (int i = 0; i < 2; i++) {
        aReg[i] = *(const uint4*)(A + (long)(m0 + aRow[i]) * K + kn + aC8[i]);
        bReg[i] = *(const uint4*)(B + (long)(kn + bRow[i]) * N + n0 + bN8[i]);
      }
    }

    // Fragment loads (batched), then 8 WMMAs
    Frag16 af[4], bfq[2];
#pragma unroll
    for (int i = 0; i < 4; i++) {
      int row = wr * 64 + i * 16 + ln;
      int k0 = hl * 8;                       // A layout: K {0..7,16..23}/{8..15,24..31}
      af[i].q[0] = *(const uint4*)(&As[row][k0]);
      af[i].q[1] = *(const uint4*)(&As[row][k0 + 16]);
    }
#pragma unroll
    for (int j = 0; j < 2; j++) {
      int col = wc * 32 + j * 16 + ln;
      int k0 = hl * 8;                       // B layout: K {0..7,16..23}/{8..15,24..31}
      bfq[j].q[0] = *(const uint4*)(&Bt[col][k0]);
      bfq[j].q[1] = *(const uint4*)(&Bt[col][k0 + 16]);
    }
#pragma unroll
    for (int i = 0; i < 4; i++)
#pragma unroll
      for (int j = 0; j < 2; j++)
        acc[i][j] = __builtin_amdgcn_wmma_f32_16x16x32_bf16(
            false, af[i].v, false, bfq[j].v, (short)0, acc[i][j], false, false);
    __syncthreads();
  }

#pragma unroll
  for (int i = 0; i < 4; i++)
#pragma unroll
    for (int j = 0; j < 2; j++)
#pragma unroll
      for (int r = 0; r < 8; r++) {
        int row = m0 + wr * 64 + i * 16 + r + 8 * hl;   // C/D layout
        int col = n0 + wc * 32 + j * 16 + ln;
        float vv = acc[i][j][r];
        if constexpr (OUT_F32)
          ((float*)Cout)[(long)row * N + col] = vv;
        else
          ((u16*)Cout)[(long)row * N + col] = f2bf(vv);
      }
}

// ---------------------------------------------------------------------------
// RoPE (reference semantics: z1=z[::2], z2=z[1::2], out=[z1*c-z2*s, z1*s+z2*c])
// ---------------------------------------------------------------------------
__global__ void rope_kernel(const u16* __restrict__ in, u16* __restrict__ out,
                            int heads, int T, long npairs) {
  long tid = (long)blockIdx.x * blockDim.x + threadIdx.x;
  if (tid >= npairs) return;
  int i = (int)(tid & 63);
  int head = (int)((tid >> 6) % heads);
  long bt = tid / ((long)64 * heads);
  int t = (int)(bt % T);
  long base = bt * (long)heads * 128 + (long)head * 128;
  float z1 = bf2f(in[base + 2 * i]);
  float z2 = bf2f(in[base + 2 * i + 1]);
  float inv = __expf(-(float)(2 * i) * (9.210340371976184f / 128.f)); // 10000^(-2i/128)
  float ang = (float)t * inv;
  float s, c;
  __sincosf(ang, &s, &c);
  out[base + i]      = f2bf(z1 * c - z2 * s);
  out[base + 64 + i] = f2bf(z1 * s + z2 * c);
}

// ---------------------------------------------------------------------------
// Causal flash attention with GQA. One block per (64-query block, b*h).
// K tile staged with GLOBAL_LOAD_ASYNC_TO_LDS_B128 (ASYNCcnt path); V tile
// batched loads + transpose scatter; Q fragments in registers; all-thread
// parallel online softmax; fragment loads batched ahead of WMMA chains.
// ---------------------------------------------------------------------------
__global__ __launch_bounds__(256) void flash_attn(const u16* __restrict__ Q,
                                                  const u16* __restrict__ Kb,
                                                  const u16* __restrict__ Vb,
                                                  u16* __restrict__ O) {
  const int T = 2048;
  __shared__ __align__(16) u16  Ks[64][136];   // [key][d]  (B for QK^T)
  __shared__ __align__(16) u16  Vt[128][72];   // [d][key]  (B for PV)
  __shared__ __align__(16) float Sb[64][68];   // f32 logits
  __shared__ __align__(16) u16  Pb[64][72];    // bf16 probabilities
  __shared__ float mbuf[64], lbuf[64], abuf[64];
  __shared__ float red[64][4];                 // softmax partial reduce

  const int tid  = threadIdx.x;
  const int lane = tid & 31;
  const int wid  = tid >> 5;
  const int hl   = lane >> 4;
  const int ln   = lane & 15;
  const int wr   = wid >> 1;    // 0..3 : 16-row strip of S / O
  const int wc   = wid & 1;     // 0..1 : 32-col strip of S, 64-col strip of O
  const int qb   = blockIdx.x;
  const int bh   = blockIdx.y;
  const int b    = bh >> 4;
  const int h    = bh & 15;
  const int kvh  = h >> 2;

  // Per-thread staging geometry (4 chunks of 8 halfs, 1024 chunks total)
  int sRow[4], sC8[4];
  unsigned kLds[4];
#pragma unroll
  for (int i = 0; i < 4; i++) {
    int idx = tid * 4 + i;
    sRow[i] = idx >> 4;
    sC8[i]  = (idx & 15) * 8;
    kLds[i] = lds_off_u(&Ks[sRow[i]][sC8[i]]);
  }

  // Q fragments for rows wr*16..+16, full d=128 (4 K-chunks of 32)
  Frag16 qf[4];
  {
    long row = (long)b * T + qb * 64 + wr * 16 + ln;
    const u16* qp = Q + row * 2048 + h * 128 + hl * 8;
#pragma unroll
    for (int c = 0; c < 4; c++) {
      qf[c].q[0] = *(const uint4*)(qp + c * 32);
      qf[c].q[1] = *(const uint4*)(qp + c * 32 + 16);
    }
  }

  v8f oacc[4];
#pragma unroll
  for (int j = 0; j < 4; j++) oacc[j] = (v8f){0.f, 0.f, 0.f, 0.f, 0.f, 0.f, 0.f, 0.f};

  if (tid < 64) { mbuf[tid] = -3.0e38f; lbuf[tid] = 0.f; }
  __syncthreads();

  for (int kb = 0; kb <= qb; kb++) {
    // ---- stage K tile [64][128] via async DMA global->LDS ----
#pragma unroll
    for (int i = 0; i < 4; i++) {
      long gr = (long)b * T + kb * 64 + sRow[i];
      const u16* gp = Kb + gr * 512 + kvh * 128 + sC8[i];
      asm volatile("global_load_async_to_lds_b128 %0, %1, off"
                   :: "v"(kLds[i]), "v"(gp)
                   : "memory");
    }
    // ---- stage V^T tile [128][64]: batched loads, then transpose scatter ----
    uint4 vreg[4];
#pragma unroll
    for (int i = 0; i < 4; i++) {
      long gr = (long)b * T + kb * 64 + sRow[i];
      vreg[i] = *(const uint4*)(Vb + gr * 512 + kvh * 128 + sC8[i]);
    }
#pragma unroll
    for (int i = 0; i < 4; i++) {
      U4E8 vv; vv.u = vreg[i];
#pragma unroll
      for (int j = 0; j < 8; j++) Vt[sC8[i] + j][sRow[i]] = vv.e[j];
    }
    // Prefetch next K/V tile toward L2 (global_prefetch_b8)
    if (kb + 1 <= qb) {
      long gr = (long)b * T + (kb + 1) * 64 + sRow[0];
      __builtin_prefetch(Kb + gr * 512 + kvh * 128, 0, 1);
      __builtin_prefetch(Vb + gr * 512 + kvh * 128, 0, 1);
    }
    asm volatile("s_wait_asynccnt 0x0" ::: "memory");   // K tile landed in LDS
    __syncthreads();

    // ---- S = Q K^T : batch all 8 B-fragment loads, then 8 WMMAs ----
    v8f sacc[2];
#pragma unroll
    for (int j = 0; j < 2; j++) sacc[j] = (v8f){0.f, 0.f, 0.f, 0.f, 0.f, 0.f, 0.f, 0.f};
    {
      Frag16 bq[8];
#pragma unroll
      for (int j = 0; j < 2; j++)
#pragma unroll
        for (int c = 0; c < 4; c++) {
          int col = wc * 32 + j * 16 + ln;
          int k0 = hl * 8;
          bq[j * 4 + c].q[0] = *(const uint4*)(&Ks[col][c * 32 + k0]);
          bq[j * 4 + c].q[1] = *(const uint4*)(&Ks[col][c * 32 + k0 + 16]);
        }
#pragma unroll
      for (int j = 0; j < 2; j++)
#pragma unroll
        for (int c = 0; c < 4; c++)
          sacc[j] = __builtin_amdgcn_wmma_f32_16x16x32_bf16(
              false, qf[c].v, false, bq[j * 4 + c].v, (short)0, sacc[j], false, false);
    }
#pragma unroll
    for (int j = 0; j < 2; j++)
#pragma unroll
      for (int r = 0; r < 8; r++)
        Sb[wr * 16 + r + 8 * hl][wc * 32 + j * 16 + ln] = sacc[j][r];
    __syncthreads();

    // ---- online softmax: 4 threads per row x 16 columns each ----
    {
      const int r = tid >> 2, s4 = tid & 3;
      const int c0 = s4 * 16;
      const int gq = qb * 64 + r, gk0 = kb * 64;
      const float sc = 0.088388347648318447f;   // 1/sqrt(128)
      float mx = -3.0e38f;
#pragma unroll
      for (int c = c0; c < c0 + 16; c++) {
        float sv = Sb[r][c] * sc;
        if (gk0 + c > gq) sv = -3.0e38f;
        mx = fmaxf(mx, sv);
      }
      red[r][s4] = mx;
      __syncthreads();
      float mold = mbuf[r];
      float m4 = fmaxf(fmaxf(fmaxf(red[r][0], red[r][1]), fmaxf(red[r][2], red[r][3])), mold);
      float alpha = __expf(mold - m4);
      float sum = 0.f;
#pragma unroll
      for (int c = c0; c < c0 + 16; c++) {
        float p = 0.f;
        if (gk0 + c <= gq) p = __expf(Sb[r][c] * sc - m4);
        sum += p;
        Pb[r][c] = f2bf(p);
      }
      __syncthreads();           // all reads of red/mbuf done
      red[r][s4] = sum;
      __syncthreads();
      if (s4 == 0) {
        mbuf[r] = m4;
        abuf[r] = alpha;
        lbuf[r] = lbuf[r] * alpha + (red[r][0] + red[r][1] + red[r][2] + red[r][3]);
      }
    }
    __syncthreads();

    // ---- rescale O, then O += P V (batched fragment loads, 8 WMMAs) ----
    float al[8];
#pragma unroll
    for (int r = 0; r < 8; r++) al[r] = abuf[wr * 16 + r + 8 * hl];
#pragma unroll
    for (int j = 0; j < 4; j++)
#pragma unroll
      for (int r = 0; r < 8; r++) oacc[j][r] *= al[r];

    {
      Frag16 pf[2], vf[8];
#pragma unroll
      for (int c = 0; c < 2; c++) {
        int prow = wr * 16 + ln;
        pf[c].q[0] = *(const uint4*)(&Pb[prow][c * 32 + hl * 8]);
        pf[c].q[1] = *(const uint4*)(&Pb[prow][c * 32 + hl * 8 + 16]);
      }
#pragma unroll
      for (int j = 0; j < 4; j++)
#pragma unroll
        for (int c = 0; c < 2; c++) {
          int col = wc * 64 + j * 16 + ln;
          vf[j * 2 + c].q[0] = *(const uint4*)(&Vt[col][c * 32 + hl * 8]);
          vf[j * 2 + c].q[1] = *(const uint4*)(&Vt[col][c * 32 + hl * 8 + 16]);
        }
#pragma unroll
      for (int j = 0; j < 4; j++)
#pragma unroll
        for (int c = 0; c < 2; c++)
          oacc[j] = __builtin_amdgcn_wmma_f32_16x16x32_bf16(
              false, pf[c].v, false, vf[j * 2 + c].v, (short)0, oacc[j], false, false);
    }
    __syncthreads();
  }

  // ---- normalize by l and write out (layout [b*T][h*128+d]) ----
  float linv[8];
#pragma unroll
  for (int r = 0; r < 8; r++) linv[r] = 1.f / lbuf[wr * 16 + r + 8 * hl];
#pragma unroll
  for (int j = 0; j < 4; j++)
#pragma unroll
    for (int r = 0; r < 8; r++) {
      long row = (long)b * T + qb * 64 + wr * 16 + r + 8 * hl;
      int col = h * 128 + wc * 64 + j * 16 + ln;
      O[row * 2048 + col] = f2bf(oacc[j][r] * linv[r]);
    }
}

// ---------------------------------------------------------------------------
// Host orchestration
// ---------------------------------------------------------------------------
extern "C" void kernel_launch(void* const* d_in, const int* in_sizes, int n_in,
                              void* d_out, int out_size, void* d_ws, size_t ws_size,
                              hipStream_t stream) {
  const int B = 4, T = 2048, DIM = 2048, NH = 16, NKV = 4;
  const long BT = (long)B * T;
  const float* x  = (const float*)d_in[0];
  const float* Wq = (const float*)d_in[1];
  const float* Wk = (const float*)d_in[2];
  const float* Wv = (const float*)d_in[3];
  const float* Wo = (const float*)d_in[4];
  float* out = (float*)d_out;

  char* ws = (char*)d_ws;
  size_t off = 0;
  auto take = [&](size_t bytes) -> char* {
    char* p = ws + off;
    off += (bytes + 255) & ~(size_t)255;
    return p;
  };
  u16* xb  = (u16*)take(BT * DIM * 2);
  u16* wqb = (u16*)take((long)DIM * DIM * 2);
  u16* wkb = (u16*)take((long)DIM * 512 * 2);
  u16* wvb = (u16*)take((long)DIM * 512 * 2);
  u16* wob = (u16*)take((long)DIM * DIM * 2);
  u16* q   = (u16*)take(BT * DIM * 2);
  u16* k   = (u16*)take(BT * 512 * 2);
  u16* v   = (u16*)take(BT * 512 * 2);
  u16* qr  = (u16*)take(BT * DIM * 2);
  u16* kr  = (u16*)take(BT * 512 * 2);
  u16* at  = (u16*)take(BT * DIM * 2);

  auto cvt = [&](const float* s, u16* d, long n) {
    long n8 = n / 8;
    cvt_f32_bf16<<<dim3((unsigned)((n8 + 255) / 256)), dim3(256), 0, stream>>>(s, d, n8);
  };
  cvt(x, xb, BT * DIM);
  cvt(Wq, wqb, (long)DIM * DIM);
  cvt(Wk, wkb, (long)DIM * 512);
  cvt(Wv, wvb, (long)DIM * 512);
  cvt(Wo, wob, (long)DIM * DIM);

  // Projections (bf16 out)
  gemm_bf16<false><<<dim3(DIM / 128, (unsigned)(BT / 128)), 256, 0, stream>>>(xb, wqb, q, (int)BT, DIM, DIM);
  gemm_bf16<false><<<dim3(512 / 128, (unsigned)(BT / 128)), 256, 0, stream>>>(xb, wkb, k, (int)BT, 512, DIM);
  gemm_bf16<false><<<dim3(512 / 128, (unsigned)(BT / 128)), 256, 0, stream>>>(xb, wvb, v, (int)BT, 512, DIM);

  // RoPE
  long nq = BT * NH * 64;
  rope_kernel<<<dim3((unsigned)((nq + 255) / 256)), 256, 0, stream>>>(q, qr, NH, T, nq);
  long nk = BT * NKV * 64;
  rope_kernel<<<dim3((unsigned)((nk + 255) / 256)), 256, 0, stream>>>(k, kr, NKV, T, nk);

  // Causal GQA flash attention
  flash_attn<<<dim3(T / 64, B * NH), 256, 0, stream>>>(qr, kr, v, at);

  // Output projection (f32 out)
  gemm_bf16<true><<<dim3(DIM / 128, (unsigned)(BT / 128)), 256, 0, stream>>>(at, wob, out, (int)BT, DIM, DIM);
}
